// MultiHeadAttention_82008105550386
// MI455X (gfx1250) — matrix operand order repaired and checked by reading, not compiled, not run on hardware
//
#include <hip/hip_runtime.h>
#include <hip/hip_bf16.h>

typedef __attribute__((ext_vector_type(16))) __bf16 v16bf;
typedef __attribute__((ext_vector_type(8)))  __bf16 v8bf;
typedef __attribute__((ext_vector_type(8)))  float  v8f;
typedef __attribute__((ext_vector_type(4)))  int    v4i;

constexpr int kB = 2, kS = 2048, kE = 1024, kH = 16, kD = 64;
constexpr int kQT = kS / 16;          // 128 query tiles per (b,h)

// ---------------- CDNA5 async global->LDS path (guarded) ----------------
#if defined(__AMDGCN__) && \
    __has_builtin(__builtin_amdgcn_global_load_async_to_lds_b128) && \
    __has_builtin(__builtin_amdgcn_s_wait_asynccnt)
#define USE_ASYNC_LDS 1
#else
#define USE_ASYNC_LDS 0
#endif

__device__ inline void cp16(const void* g, void* l) {
#if USE_ASYNC_LDS
  __builtin_amdgcn_global_load_async_to_lds_b128(
      (__attribute__((address_space(1))) v4i*)(v4i*)(g),
      (__attribute__((address_space(3))) v4i*)(v4i*)(l), 0, 0);
#else
  *(uint4*)l = *(const uint4*)g;      // global_load_b128 + ds_store_b128
#endif
}

__device__ inline void async_drain() {
#if USE_ASYNC_LDS
  __builtin_amdgcn_s_wait_asynccnt(0);
#endif
}

// ---- A-fragment (16x32 bf16) loader: lane holds K = 16*hh+{0..15}
__device__ inline v16bf load_a32(const __bf16* rowp, int hh) {
  v8bf lo = *(const v8bf*)(rowp + 16 * hh);
  v8bf hi = *(const v8bf*)(rowp + 16 * hh + 8);
  v16bf a;
#pragma unroll
  for (int i = 0; i < 8; ++i) { a[i] = lo[i]; a[i + 8] = hi[i]; }
  return a;
}

// =====================================================================
// Stage 1: W[h][e][d] f32  ->  WT[h][d][e] bf16 (transpose + downconvert)
// =====================================================================
__global__ void wtrans_kernel(const float* __restrict__ W, __bf16* __restrict__ WT) {
  size_t idx = (size_t)blockIdx.x * blockDim.x + threadIdx.x;
  if (idx >= (size_t)kH * kE * kD) return;
  int d = (int)(idx % kD);
  int e = (int)((idx / kD) % kE);
  int h = (int)(idx / ((size_t)kD * kE));
  WT[((size_t)h * kD + d) * kE + e] = (__bf16)W[idx];
}

// =====================================================================
// Stage 2: per-wave 16x64 projection tile:  X[b, s0:s0+16, :] * W[h]
//   transpose_out==0 : OUT bf16 [b][h][s][d]   (for Q, K)
//   transpose_out==1 : OUT bf16 [b][h][d][s]   (for V)
// =====================================================================
__global__ void __launch_bounds__(128)
proj_kernel(const float* __restrict__ X, const __bf16* __restrict__ WT,
            __bf16* __restrict__ OUT, int transpose_out) {
  int lane = threadIdx.x & 31;
  int wid  = threadIdx.x >> 5;
  int gid  = blockIdx.x * 4 + wid;          // 4096 waves total
  int qt = gid % kQT;
  int h  = (gid / kQT) % kH;
  int b  = gid / (kQT * kH);
  int n  = lane & 15;                        // A-row / B-col
  int hh = lane >> 4;
  int s0 = qt * 16;

  const float*  xrow = X  + ((size_t)b * kS + s0 + n) * kE;
  const __bf16* wt   = WT + (size_t)h * kD * kE;

  v8f acc[4] = {};
  for (int e0 = 0; e0 < kE; e0 += 32) {
    v16bf a;
    const float* p1 = xrow + e0 + 16 * hh;
    const float* p2 = xrow + e0 + 16 * hh + 8;
#pragma unroll
    for (int i = 0; i < 8; ++i) { a[i] = (__bf16)p1[i]; a[i + 8] = (__bf16)p2[i]; }
#pragma unroll
    for (int t = 0; t < 4; ++t) {
      v16bf bf = *(const v16bf*)(wt + (size_t)(t * 16 + n) * kE + e0 + 16 * hh);
      acc[t] = __builtin_amdgcn_wmma_f32_16x16x32_bf16(
          false, a, false, bf, (short)0, acc[t], false, false);
    }
  }

  __bf16* ob = OUT + ((size_t)b * kH + h) * (size_t)kS * kD;
  if (!transpose_out) {
#pragma unroll
    for (int t = 0; t < 4; ++t)
#pragma unroll
      for (int i = 0; i < 8; ++i)
        ob[(size_t)(s0 + i + 8 * hh) * kD + t * 16 + n] = (__bf16)acc[t][i];
  } else {
#pragma unroll
    for (int t = 0; t < 4; ++t)
#pragma unroll
      for (int i = 0; i < 8; ++i)
        ob[(size_t)(t * 16 + n) * kS + s0 + i + 8 * hh] = (__bf16)acc[t][i];
  }
}

// =====================================================================
// Stage 3: flash attention, 4 waves/block share K/V tiles via LDS with
// double-buffered async global->LDS copies. Wave w handles q-rows
// [qbase+16w, qbase+16w+16); k-loop bound is uniform over the block.
//   Q,K bf16 [b][h][s][d] ; VT bf16 [b][h][d][s] ; OUT f32 [b][s][h*d]
// =====================================================================
__device__ inline void stage_kv(const __bf16* kb, const __bf16* vb, int k0,
                                __bf16* kdst, __bf16* vdst, int tid) {
  // K tile: 32 keys x 64 d = 4KB, contiguous in global ([s][d] row-major)
  const char* kg = (const char*)(kb + (size_t)k0 * kD);
  char* kl = (char*)kdst;
#pragma unroll
  for (int r = 0; r < 2; ++r) {
    int c = tid + 128 * r;                       // 256 x 16B chunks
    cp16(kg + (size_t)c * 16, kl + c * 16);
  }
  // V tile: 64 d-rows x 32 keys = 4KB, 64B per row strided by kS*2 bytes
  char* vl = (char*)vdst;
#pragma unroll
  for (int r = 0; r < 2; ++r) {
    int c = tid + 128 * r;                       // 256 x 16B chunks
    int d = c >> 2, part = c & 3;
    cp16((const char*)(vb + (size_t)d * kS + k0) + part * 16, vl + c * 16);
  }
}

__global__ void __launch_bounds__(128)
attn_kernel(const __bf16* __restrict__ Q, const __bf16* __restrict__ K,
            const __bf16* __restrict__ VT, float* __restrict__ OUT) {
  __shared__ __align__(16) __bf16 Kb[2][32 * 64];   // [key][d], double buffered
  __shared__ __align__(16) __bf16 Vb[2][64 * 32];   // [d][key], double buffered
  __shared__ __align__(16) __bf16 Pb[4][16 * 32];   // per-wave P staging

  int tid  = threadIdx.x;
  int lane = tid & 31;
  int wid  = tid >> 5;
  const int NBT = kS / 64;                          // 32 q-blocks per (b,h)
  int bt = blockIdx.x % NBT;
  int h  = (blockIdx.x / NBT) % kH;
  int b  = blockIdx.x / (NBT * kH);
  int n  = lane & 15;
  int hh = lane >> 4;
  int qbase = bt * 64;
  int q0 = qbase + wid * 16;
  int kmax = qbase + 64;                            // uniform over block

  const __bf16* qb = Q  + ((size_t)b * kH + h) * (size_t)kS * kD;
  const __bf16* kb = K  + ((size_t)b * kH + h) * (size_t)kS * kD;
  const __bf16* vb = VT + ((size_t)b * kH + h) * (size_t)kD * kS;

  // Q A-fragments, held in VGPRs for the whole k-loop
  const __bf16* qrow = qb + (size_t)(q0 + n) * kD;
  v16bf qa0 = load_a32(qrow, hh);        // d = 0..31
  v16bf qa1 = load_a32(qrow + 32, hh);   // d = 32..63

  float m[8], l[8];
#pragma unroll
  for (int i = 0; i < 8; ++i) { m[i] = -1e30f; l[i] = 0.f; }
  v8f acc[4] = {};

  __bf16* pb = &Pb[wid][0];
  const float scale = 0.125f;            // 64^-0.5

  stage_kv(kb, vb, 0, Kb[0], Vb[0], tid);          // prologue fill buffer 0

  for (int k0 = 0; k0 < kmax; k0 += 32) {
    int bi = (k0 >> 5) & 1;
    async_drain();                                  // this wave's copies done
    __syncthreads();                                // tile visible, prev compute done
    if (k0 + 32 < kmax)                             // prefetch next tile
      stage_kv(kb, vb, k0 + 32, Kb[bi ^ 1], Vb[bi ^ 1], tid);

    if (k0 < q0 + 16) {                             // wave-uniform causal predicate
      const __bf16* kt = &Kb[bi][0];
      const __bf16* vt = &Vb[bi][0];

      // ---- S = Q K^T for key columns [k0, k0+32)
      v8f st[2] = {};
#pragma unroll
      for (int t = 0; t < 2; ++t) {
        const __bf16* krow = kt + (size_t)(t * 16 + n) * kD;
        v16bf b0 = *(const v16bf*)(krow + 16 * hh);       // d 0..31
        v16bf b1 = *(const v16bf*)(krow + 32 + 16 * hh);  // d 32..63
        st[t] = __builtin_amdgcn_wmma_f32_16x16x32_bf16(
            false, qa0, false, b0, (short)0, st[t], false, false);
        st[t] = __builtin_amdgcn_wmma_f32_16x16x32_bf16(
            false, qa1, false, b1, (short)0, st[t], false, false);
      }

      // ---- scale + causal mask (boundary blocks only)
      bool needmask = (k0 + 32 > q0);
#pragma unroll
      for (int t = 0; t < 2; ++t)
#pragma unroll
        for (int i = 0; i < 8; ++i) {
          float s = st[t][i] * scale;
          if (needmask && (k0 + t * 16 + n) > (q0 + i + 8 * hh)) s = -1e30f;
          st[t][i] = s;
        }

      // ---- online softmax (row = i + 8*hh; 16-lane reductions)
#pragma unroll
      for (int i = 0; i < 8; ++i) {
        float mx = fmaxf(st[0][i], st[1][i]);
        for (int off = 1; off < 16; off <<= 1)
          mx = fmaxf(mx, __shfl_xor(mx, off, 32));
        float mn = fmaxf(m[i], mx);
        float alpha = __expf(m[i] - mn);
        float p0 = __expf(st[0][i] - mn);
        float p1 = __expf(st[1][i] - mn);
        float sum = p0 + p1;
        for (int off = 1; off < 16; off <<= 1)
          sum += __shfl_xor(sum, off, 32);
        l[i] = l[i] * alpha + sum;
        m[i] = mn;
#pragma unroll
        for (int t = 0; t < 4; ++t) acc[t][i] *= alpha;
        int rowm = i + 8 * hh;
        pb[rowm * 32 + n]      = (__bf16)p0;        // C-layout -> LDS
        pb[rowm * 32 + 16 + n] = (__bf16)p1;
      }
      __builtin_amdgcn_wave_barrier();              // LDS in-order within wave

      // ---- O += P * V  (P re-read in A-layout; Vb rows are B-frags)
      v16bf pa = load_a32(pb + n * 32, hh);
#pragma unroll
      for (int t = 0; t < 4; ++t) {
        v16bf bv = *(const v16bf*)(vt + (size_t)(t * 16 + n) * 32 + 16 * hh);
        acc[t] = __builtin_amdgcn_wmma_f32_16x16x32_bf16(
            false, pa, false, bv, (short)0, acc[t], false, false);
      }
      __builtin_amdgcn_wave_barrier();
    }
  }

  // ---- epilogue: normalize, write fp32 out[b][s][h*64 + d]
  float* ob = OUT + (size_t)b * kS * (kH * kD) + (size_t)h * kD;
#pragma unroll
  for (int t = 0; t < 4; ++t)
#pragma unroll
    for (int i = 0; i < 8; ++i)
      ob[(size_t)(q0 + i + 8 * hh) * (kH * kD) + t * 16 + n] = acc[t][i] / l[i];
}

// =====================================================================
extern "C" void kernel_launch(void* const* d_in, const int* in_sizes, int n_in,
                              void* d_out, int out_size, void* d_ws, size_t ws_size,
                              hipStream_t stream) {
  const float* query = (const float*)d_in[0];
  const float* key   = (const float*)d_in[1];
  const float* value = (const float*)d_in[2];
  /* d_in[3] = causal mask, handled analytically */
  const float* Wq = (const float*)d_in[4];
  const float* Wk = (const float*)d_in[5];
  const float* Wv = (const float*)d_in[6];

  char* ws = (char*)d_ws;                 // 30 MB layout
  __bf16* WqT = (__bf16*)(ws + (0ull  << 20));
  __bf16* WkT = (__bf16*)(ws + (2ull  << 20));
  __bf16* WvT = (__bf16*)(ws + (4ull  << 20));
  __bf16* qws = (__bf16*)(ws + (6ull  << 20));
  __bf16* kws = (__bf16*)(ws + (14ull << 20));
  __bf16* vtw = (__bf16*)(ws + (22ull << 20));

  int welems = kH * kE * kD;
  wtrans_kernel<<<(welems + 255) / 256, 256, 0, stream>>>(Wq, WqT);
  wtrans_kernel<<<(welems + 255) / 256, 256, 0, stream>>>(Wk, WkT);
  wtrans_kernel<<<(welems + 255) / 256, 256, 0, stream>>>(Wv, WvT);

  int pblocks = (kB * kH * kQT) / 4;      // 1024 blocks * 4 waves
  proj_kernel<<<pblocks, 128, 0, stream>>>(query, WqT, qws, 0);
  proj_kernel<<<pblocks, 128, 0, stream>>>(key,   WkT, kws, 0);
  proj_kernel<<<pblocks, 128, 0, stream>>>(value, WvT, vtw, 1);

  int ablocks = kB * kH * (kS / 64);      // 1024 blocks * 4 waves
  attn_kernel<<<ablocks, 128, 0, stream>>>(qws, kws, vtw, (float*)d_out);
}
